// SelfCorrelationPercPooling_16638703305475
// MI455X (gfx1250) — matrix operand order repaired and checked by reading, compiled, not run on hardware
//
#include <hip/hip_runtime.h>
#include <hip/hip_bf16.h>
#include <math.h>

// ---------------------------------------------------------------------------
// SelfCorrelationPercPooling for MI455X (gfx1250, wave32, WMMA)
//   x: [8, 48, 48, 256] f32  ->  out: [8, 48, 48, 115] f32
// Pipeline:
//   1) normalize_kernel : per-pixel channel mean/std -> Xn as f16 (ws)
//   2) corr_wmma_kernel : corr[b] = Xn Xn^T / 256 via v_wmma_f32_16x16x32_f16
//   3) sort_pick_kernel : bitonic sort (desc) each 2304-row in LDS, gather 115 ranks
// Workspace: [0, 9437184) f16 Xn ; [9437184, ~179.4MB) f32 corr
// ---------------------------------------------------------------------------

typedef _Float16 h8  __attribute__((ext_vector_type(8)));
typedef _Float16 h16 __attribute__((ext_vector_type(16)));
typedef float    f8  __attribute__((ext_vector_type(8)));

#define NMAPS   2304     // 48*48
#define NCH     256
#define NPOOLS  115
#define TPD     144      // 2304/16 tiles per dim
#define SORT_N  4096

// ---------------------------------------------------------------------------
// Kernel 1: channel-normalize, one wave32 per pixel (8 pixels / 256-thr block)
// ---------------------------------------------------------------------------
__global__ void normalize_kernel(const float* __restrict__ x,
                                 _Float16* __restrict__ Xn) {
    const int lane = threadIdx.x & 31;
    const int wave = threadIdx.x >> 5;
    const int pix  = blockIdx.x * 8 + wave;          // 0 .. 18431

    const float* __restrict__ xp = x + (size_t)pix * NCH + lane * 8;
    float4 v0 = *(const float4*)(xp);
    float4 v1 = *(const float4*)(xp + 4);

    float s  = v0.x + v0.y + v0.z + v0.w + v1.x + v1.y + v1.z + v1.w;
    float ss = v0.x*v0.x + v0.y*v0.y + v0.z*v0.z + v0.w*v0.w
             + v1.x*v1.x + v1.y*v1.y + v1.z*v1.z + v1.w*v1.w;

    // wave32 butterfly reduction
    #pragma unroll
    for (int off = 16; off > 0; off >>= 1) {
        s  += __shfl_xor(s,  off, 32);
        ss += __shfl_xor(ss, off, 32);
    }

    const float mean = s * (1.0f / NCH);
    float var  = ss * (1.0f / NCH) - mean * mean;
    var = fmaxf(var, 0.0f);
    const float inv = 1.0f / fmaxf(1e-4f, sqrtf(var));

    h8 o;
    o[0] = (_Float16)((v0.x - mean) * inv);
    o[1] = (_Float16)((v0.y - mean) * inv);
    o[2] = (_Float16)((v0.z - mean) * inv);
    o[3] = (_Float16)((v0.w - mean) * inv);
    o[4] = (_Float16)((v1.x - mean) * inv);
    o[5] = (_Float16)((v1.y - mean) * inv);
    o[6] = (_Float16)((v1.z - mean) * inv);
    o[7] = (_Float16)((v1.w - mean) * inv);
    *(h8*)(Xn + (size_t)pix * NCH + lane * 8) = o;
}

// ---------------------------------------------------------------------------
// Kernel 2: corr = Xn Xn^T / 256, one wave32 per 16x16 tile, K=256 in 8 steps
// A-operand (16x32 f16) lane layout per CDNA5 ISA:
//   lane L (0..31): M = L&15 ; lanes>=16 add K+8
//   element e (0..15): K = kBase + (L>>4)*8 + (e&7) + (e>>3)*16
// B-operand (32x16 f16): lane L: N = L&15 ; K = kBase + (L>>4)*8 + (e&7) + (e>>3)*16
// C/D (16x16 f32, 8 VGPRs): lane L: N = L&15 ; vgpr r: M = r + (L>>4)*8
// ---------------------------------------------------------------------------
__global__ void corr_wmma_kernel(const _Float16* __restrict__ Xn,
                                 float* __restrict__ corr) {
    const int lane   = threadIdx.x & 31;
    const int waveId = threadIdx.x >> 5;
    const int tile   = blockIdx.x * 8 + waveId;      // 0 .. 165887

    const int b  = tile / (TPD * TPD);
    const int t  = tile % (TPD * TPD);
    const int mt = t / TPD;
    const int nt = t % TPD;
    const int rowBase = mt * 16;
    const int colBase = nt * 16;

    const _Float16* __restrict__ Ab = Xn + (size_t)b * NMAPS * NCH;
    const _Float16* __restrict__ arow = Ab + (size_t)(rowBase + (lane & 15)) * NCH;
    const _Float16* __restrict__ brow = Ab + (size_t)(colBase + (lane & 15)) * NCH;
    const int aoff = (lane >> 4) << 3;   // 0 or 8
    const int boff = (lane >> 4) << 3;   // 0 or 8

    f8 acc = {};

    #pragma unroll
    for (int k = 0; k < NCH; k += 32) {
        h8 a0 = *(const h8*)(arow + k + aoff);        // K = kb + {0..7}(+8)
        h8 a1 = *(const h8*)(arow + k + aoff + 16);   // K = kb + {16..23}(+8)
        h8 b0 = *(const h8*)(brow + k + boff);        // K = kb + {0..7}(+8)
        h8 b1 = *(const h8*)(brow + k + boff + 16);   // K = kb + {16..23}(+8)

        h16 av = __builtin_shufflevector(a0, a1, 0,1,2,3,4,5,6,7,
                                                 8,9,10,11,12,13,14,15);
        h16 bv = __builtin_shufflevector(b0, b1, 0,1,2,3,4,5,6,7,
                                                 8,9,10,11,12,13,14,15);

        acc = __builtin_amdgcn_wmma_f32_16x16x32_f16(
                  /*neg_a=*/false, av, /*neg_b=*/false, bv,
                  /*c_mod=*/(short)0, acc, /*reuse_a=*/false, /*reuse_b=*/false);
    }

    const int n    = colBase + (lane & 15);
    const int mOff = (lane >> 4) << 3;
    float* __restrict__ outB = corr + (size_t)b * NMAPS * NMAPS;
    #pragma unroll
    for (int r = 0; r < 8; ++r) {
        const int m = rowBase + mOff + r;
        outB[(size_t)m * NMAPS + n] = acc[r] * (1.0f / NCH);
    }
}

// ---------------------------------------------------------------------------
// Kernel 3: descending bitonic sort of one 2304-row (padded to 4096 with -inf)
// in 16KB LDS, then gather the 115 percentile ranks.
// ---------------------------------------------------------------------------
__global__ void sort_pick_kernel(const float* __restrict__ corr,
                                 float* __restrict__ out) {
    __shared__ float buf[SORT_N];
    const int row = blockIdx.x;                       // 0 .. 18431
    const float* __restrict__ src = corr + (size_t)row * NMAPS;

    for (int i = threadIdx.x; i < SORT_N; i += blockDim.x)
        buf[i] = (i < NMAPS) ? src[i] : -INFINITY;
    __syncthreads();

    for (int k = 2; k <= SORT_N; k <<= 1) {
        for (int j = k >> 1; j > 0; j >>= 1) {
            for (int i = threadIdx.x; i < SORT_N; i += blockDim.x) {
                const int ixj = i ^ j;
                if (ixj > i) {
                    const float a = buf[i];
                    const float c = buf[ixj];
                    const bool descRegion = ((i & k) == 0);
                    if (descRegion ? (a < c) : (a > c)) {
                        buf[i]   = c;
                        buf[ixj] = a;
                    }
                }
            }
            __syncthreads();
        }
    }

    // ranks = round(linspace(1, 2303, 115)); step = 2302/114, no exact .5 ties
    for (int jj = threadIdx.x; jj < NPOOLS; jj += blockDim.x) {
        const float pos = 1.0f + (float)jj * (2302.0f / 114.0f);
        const int r = (int)(pos + 0.5f);
        out[(size_t)row * NPOOLS + jj] = buf[r];
    }
}

// ---------------------------------------------------------------------------
extern "C" void kernel_launch(void* const* d_in, const int* in_sizes, int n_in,
                              void* d_out, int out_size, void* d_ws, size_t ws_size,
                              hipStream_t stream) {
    (void)in_sizes; (void)n_in; (void)out_size; (void)ws_size;

    const float* x   = (const float*)d_in[0];
    float*       out = (float*)d_out;

    // ws layout: [0, 9437184) = Xn f16 (8*2304*256*2B)
    //            [9437184, +169.87MB) = corr f32 (8*2304*2304*4B)
    _Float16* Xn   = (_Float16*)d_ws;
    float*    corr = (float*)((char*)d_ws + (size_t)9437184);

    // 18432 pixels, 8 per block
    normalize_kernel<<<NMAPS, 256, 0, stream>>>(x, Xn);

    // 8 * 144 * 144 = 165888 tiles, 8 waves (tiles) per 256-thread block
    corr_wmma_kernel<<<165888 / 8, 256, 0, stream>>>(Xn, corr);

    // one row per block
    sort_pick_kernel<<<8 * NMAPS, 256, 0, stream>>>(corr, out);
}
